// EncoderLayer_62491774156985
// MI455X (gfx1250) — matrix operand order repaired and checked by reading, not compiled, not run on hardware
//
#include <hip/hip_runtime.h>

// ---------------------------------------------------------------------------
// MI455X (gfx1250) encoder layer: f16 WMMA GEMMs (double-buffered, async
// global->LDS staging) + flash attention + fp32 LayerNorm.
// ---------------------------------------------------------------------------

typedef __attribute__((ext_vector_type(16))) _Float16 v16h;
typedef __attribute__((ext_vector_type(8)))  _Float16 v8h;
typedef __attribute__((ext_vector_type(8)))  float    v8f;

union AFrag { v16h v; v8h h[2]; };

#define WMMA16(A, B, C) \
  __builtin_amdgcn_wmma_f32_16x16x32_f16(false, (A), false, (B), (short)0, (C), false, false)

// async copy of 16 bytes global -> LDS (per lane), tracked by ASYNCcnt
__device__ __forceinline__ void async_b128(const void* gptr, void* lptr) {
  unsigned lds = (unsigned)(unsigned long long)lptr;  // LDS offset = low 32 bits
  asm volatile("global_load_async_to_lds_b128 %0, %1, off"
               :: "v"(lds), "v"(gptr) : "memory");
}
__device__ __forceinline__ void wait_async0() {
  asm volatile("s_wait_asynccnt 0x0" ::: "memory");
}

// ------------------------------- convert -----------------------------------
__global__ __launch_bounds__(256) void f32_to_f16_kernel(
    const float* __restrict__ in, _Float16* __restrict__ out, int n) {
  int i = blockIdx.x * 256 + threadIdx.x;
  if (i < n) out[i] = (_Float16)in[i];
}

// ------------------------------- GEMM --------------------------------------
// C[M,N] = A[M,K](f16,row-major) * B[K,N](f16,row-major)  (+bias)(+relu)
// Block tile 128x128, BK=32, 8 waves in 2(M)x4(N) grid, wave tile 64x32.
// Double-buffered LDS; tile k+1 prefetched via async-to-LDS during compute.
template <bool BIAS, bool RELU, bool F16OUT>
__global__ __launch_bounds__(256) void gemm_wmma(
    const _Float16* __restrict__ A, const _Float16* __restrict__ Bw,
    const float* __restrict__ bias, float* __restrict__ Cf,
    _Float16* __restrict__ Ch, int M, int N, int K) {
  __shared__ __align__(16) _Float16 Als[2][128][40];   // 128x32, pad->40
  __shared__ __align__(16) _Float16 Bls[2][32][136];   // 32x128, pad->136

  const int m0 = blockIdx.y * 128;
  const int n0 = blockIdx.x * 128;
  const int t = threadIdx.x;
  const int wave = t >> 5, lane = t & 31;
  const int wm = wave >> 2;        // 0..1  -> rows wm*64
  const int wn = wave & 3;         // 0..3  -> cols wn*32
  const int mlane = lane & 15;
  const int khalf = (lane >> 4) * 8;   // 0 for lanes 0-15, 8 for lanes 16-31

  // per-thread staging coordinates (2 chunks of 16B for A, 2 for B)
  const int ar0 = t >> 2,        as0 = (t & 3) * 8;         // A chunk 0
  const int ar1 = (t + 256) >> 2, as1 = ((t + 256) & 3) * 8; // A chunk 1
  const int br0 = t >> 4,        bs0 = (t & 15) * 8;        // B chunk 0
  const int br1 = (t + 256) >> 4, bs1 = ((t + 256) & 15) * 8;// B chunk 1

  auto stage = [&](int buf, int k0) {
    async_b128(&A[(size_t)(m0 + ar0) * K + k0 + as0], &Als[buf][ar0][as0]);
    async_b128(&A[(size_t)(m0 + ar1) * K + k0 + as1], &Als[buf][ar1][as1]);
    async_b128(&Bw[(size_t)(k0 + br0) * N + n0 + bs0], &Bls[buf][br0][bs0]);
    async_b128(&Bw[(size_t)(k0 + br1) * N + n0 + bs1], &Bls[buf][br1][bs1]);
  };

  v8f acc[4][2] = {};

  stage(0, 0);
  wait_async0();
  __syncthreads();

  int buf = 0;
  for (int k0 = 0; k0 < K; k0 += 32) {
    if (k0 + 32 < K) stage(buf ^ 1, k0 + 32);   // prefetch next tile (async)

    AFrag a[4], b[2];
#pragma unroll
    for (int mf = 0; mf < 4; mf++) {
      int mr = wm * 64 + mf * 16 + mlane;
      a[mf].h[0] = *(const v8h*)&Als[buf][mr][khalf];       // K 0..7 / 8..15
      a[mf].h[1] = *(const v8h*)&Als[buf][mr][16 + khalf];  // K 16..23 / 24..31
    }
#pragma unroll
    for (int nf = 0; nf < 2; nf++) {
      int nc = wn * 32 + nf * 16;
      for (int s = 0; s < 8; s++) {                         // lane = N col, K table order
        b[nf].h[0][s] = Bls[buf][khalf + s][nc + mlane];
        b[nf].h[1][s] = Bls[buf][16 + khalf + s][nc + mlane];
      }
    }
#pragma unroll
    for (int mf = 0; mf < 4; mf++)
#pragma unroll
      for (int nf = 0; nf < 2; nf++)
        acc[mf][nf] = WMMA16(a[mf].v, b[nf].v, acc[mf][nf]);

    wait_async0();        // own prefetch complete
    __syncthreads();      // everyone done reading buf / writing buf^1
    buf ^= 1;
  }

  // epilogue: C layout lane=N (lane&15), M = vgpr + 8*(lane>=16)
#pragma unroll
  for (int mf = 0; mf < 4; mf++)
#pragma unroll
    for (int nf = 0; nf < 2; nf++) {
      int n = n0 + wn * 32 + nf * 16 + mlane;
      float bv = BIAS ? bias[n] : 0.0f;
#pragma unroll
      for (int v = 0; v < 8; v++) {
        int m = m0 + wm * 64 + mf * 16 + v + khalf;
        float val = acc[mf][nf][v] + bv;
        if (RELU) val = fmaxf(val, 0.0f);
        if (F16OUT) Ch[(size_t)m * N + n] = (_Float16)val;
        else        Cf[(size_t)m * N + n] = val;
      }
    }
}

// --------------------------- flash attention --------------------------------
// One block: 128 q rows of one (b,h). K chunks of 64. Online softmax (fp32
// running max/sum per row), scores/probs staged in LDS as f16, PV via WMMA.
// Q and V tiles staged with async-to-LDS; K staged via registers (transpose).
__global__ __launch_bounds__(256) void attention_kernel(
    const _Float16* __restrict__ Qg, const _Float16* __restrict__ Kg,
    const _Float16* __restrict__ Vg, const int* __restrict__ mask,
    _Float16* __restrict__ Ctx) {
  const int S = 2048, D = 1024;
  __shared__ __align__(16) _Float16 Qs[128][72];   // q tile [128 x 64]
  __shared__ __align__(16) _Float16 Kts[64][72];   // K^T chunk [d=64 x kc=64]
  __shared__ __align__(16) _Float16 Vs[64][72];    // V chunk  [kc=64 x dv=64]
  __shared__ __align__(16) _Float16 Sp[128][72];   // scores, then probs (f16)
  __shared__ float rowf[128];
  __shared__ float rowl[128];

  const int t = threadIdx.x;
  const int wave = t >> 5, lane = t & 31;
  const int wm = wave >> 1;        // 0..3 -> rows wm*32
  const int wn = wave & 1;         // 0..1 -> cols wn*32
  const int mlane = lane & 15;
  const int khalf = (lane >> 4) * 8;

  const int q0 = blockIdx.x * 128;
  const int bh = blockIdx.y;
  const int b = bh >> 4, h = bh & 15;
  const size_t rowbase = (size_t)b * S;
  const int colbase = h * 64;
  const int* mrow = mask + b * S;

  // stage Q tile: 1024 x 16B async chunks
#pragma unroll
  for (int i = 0; i < 4; i++) {
    int idx = t + i * 256;
    int r = idx >> 3, s = (idx & 7) * 8;
    async_b128(&Qg[(rowbase + q0 + r) * D + colbase + s], &Qs[r][s]);
  }

  v8f acc[2][2] = {};
  float mold = -3.0e38f, lrun = 0.0f;

  for (int kc = 0; kc < S; kc += 64) {
    // stage V chunk (async) and K^T chunk (transpose on LDS write)
#pragma unroll
    for (int i = 0; i < 2; i++) {
      int idx = t + i * 256;
      int r = idx >> 3, s = (idx & 7) * 8;
      async_b128(&Vg[(rowbase + kc + r) * D + colbase + s], &Vs[r][s]);
      v8h kk = *(const v8h*)&Kg[(rowbase + kc + r) * D + colbase + s];
#pragma unroll
      for (int j = 0; j < 8; j++) Kts[s + j][r] = kk[j];
    }
    wait_async0();
    __syncthreads();

    // S = Q * K^T  (M=128 q, N=64 k-cols, K=64 d)
    v8f sacc[2][2] = {};
#pragma unroll
    for (int d0 = 0; d0 < 64; d0 += 32) {
      AFrag a[2], bf[2];
#pragma unroll
      for (int mf = 0; mf < 2; mf++) {
        int mr = wm * 32 + mf * 16 + mlane;
        a[mf].h[0] = *(const v8h*)&Qs[mr][d0 + khalf];
        a[mf].h[1] = *(const v8h*)&Qs[mr][d0 + 16 + khalf];
      }
#pragma unroll
      for (int nf = 0; nf < 2; nf++) {
        int nc = wn * 32 + nf * 16;
        for (int s = 0; s < 8; s++) {
          bf[nf].h[0][s] = Kts[d0 + khalf + s][nc + mlane];
          bf[nf].h[1][s] = Kts[d0 + 16 + khalf + s][nc + mlane];
        }
      }
#pragma unroll
      for (int mf = 0; mf < 2; mf++)
#pragma unroll
        for (int nf = 0; nf < 2; nf++)
          sacc[mf][nf] = WMMA16(a[mf].v, bf[nf].v, sacc[mf][nf]);
    }
    // spill scaled scores to LDS (f16)
#pragma unroll
    for (int mf = 0; mf < 2; mf++)
#pragma unroll
      for (int nf = 0; nf < 2; nf++) {
        int c = wn * 32 + nf * 16 + mlane;
#pragma unroll
        for (int v = 0; v < 8; v++) {
          int r = wm * 32 + mf * 16 + v + khalf;
          Sp[r][c] = (_Float16)(sacc[mf][nf][v] * 0.125f);
        }
      }
    __syncthreads();

    // online softmax: thread t<128 owns row t (in-place s -> p)
    if (t < 128) {
      float cm = -3.0e38f;
      for (int j = 0; j < 64; j++) {
        float s = (float)Sp[t][j];
        if (mrow[kc + j] == 0) s = -1.0e9f;
        cm = fmaxf(cm, s);
      }
      float mnew = fmaxf(mold, cm);
      float f = __expf(mold - mnew);
      float sum = 0.0f;
      for (int j = 0; j < 64; j++) {
        float s = (float)Sp[t][j];
        if (mrow[kc + j] == 0) s = -1.0e9f;
        float p = __expf(s - mnew);
        Sp[t][j] = (_Float16)p;
        sum += p;
      }
      lrun = lrun * f + sum;
      rowf[t] = f;
      mold = mnew;
    }
    __syncthreads();

    // rescale accumulators, then acc += P * V  (K dim = 64 kc rows)
#pragma unroll
    for (int mf = 0; mf < 2; mf++)
#pragma unroll
      for (int v = 0; v < 8; v++) {
        int r = wm * 32 + mf * 16 + v + khalf;
        float f = rowf[r];
        acc[mf][0][v] *= f;
        acc[mf][1][v] *= f;
      }
#pragma unroll
    for (int d0 = 0; d0 < 64; d0 += 32) {
      AFrag a[2], bf[2];
#pragma unroll
      for (int mf = 0; mf < 2; mf++) {
        int mr = wm * 32 + mf * 16 + mlane;
        a[mf].h[0] = *(const v8h*)&Sp[mr][d0 + khalf];
        a[mf].h[1] = *(const v8h*)&Sp[mr][d0 + 16 + khalf];
      }
#pragma unroll
      for (int nf = 0; nf < 2; nf++) {
        int nc = wn * 32 + nf * 16;
        for (int s = 0; s < 8; s++) {
          bf[nf].h[0][s] = Vs[d0 + khalf + s][nc + mlane];
          bf[nf].h[1][s] = Vs[d0 + 16 + khalf + s][nc + mlane];
        }
      }
#pragma unroll
      for (int mf = 0; mf < 2; mf++)
#pragma unroll
        for (int nf = 0; nf < 2; nf++)
          acc[mf][nf] = WMMA16(a[mf].v, bf[nf].v, acc[mf][nf]);
    }
    __syncthreads();
  }

  if (t < 128) rowl[t] = lrun;
  __syncthreads();

  // normalize and write ctx (f16, [B*S, H*64] layout)
#pragma unroll
  for (int mf = 0; mf < 2; mf++)
#pragma unroll
    for (int nf = 0; nf < 2; nf++) {
      int c = wn * 32 + nf * 16 + mlane;
#pragma unroll
      for (int v = 0; v < 8; v++) {
        int r = wm * 32 + mf * 16 + v + khalf;
        float val = acc[mf][nf][v] / rowl[r];
        Ctx[(rowbase + q0 + r) * D + colbase + c] = (_Float16)val;
      }
    }
}

// ---------------------- residual + LayerNorm (ddof=1) -----------------------
template <bool F16OUT>
__global__ __launch_bounds__(256) void ln_residual_kernel(
    const float* __restrict__ X, const float* __restrict__ R,
    const float* __restrict__ alpha, const float* __restrict__ beta,
    float* __restrict__ Y, _Float16* __restrict__ Yh) {
  const int D = 1024;
  __shared__ float red[256];
  const int row = blockIdx.x;
  const int t = threadIdx.x;
  const float* xr = X + (size_t)row * D;
  const float* rr = R + (size_t)row * D;

  float v[4];
  float s = 0.0f;
#pragma unroll
  for (int i = 0; i < 4; i++) {
    int c = t + i * 256;
    v[i] = xr[c] + rr[c];
    s += v[i];
  }
  red[t] = s;
  __syncthreads();
  for (int o = 128; o > 0; o >>= 1) {
    if (t < o) red[t] += red[t + o];
    __syncthreads();
  }
  float mean = red[0] * (1.0f / D);
  __syncthreads();

  float ss = 0.0f;
#pragma unroll
  for (int i = 0; i < 4; i++) {
    float d = v[i] - mean;
    ss += d * d;
  }
  red[t] = ss;
  __syncthreads();
  for (int o = 128; o > 0; o >>= 1) {
    if (t < o) red[t] += red[t + o];
    __syncthreads();
  }
  float var = red[0] * (1.0f / (D - 1));       // Bessel correction (torch.std)
  float inv = 1.0f / (sqrtf(var) + 1e-6f);

#pragma unroll
  for (int i = 0; i < 4; i++) {
    int c = t + i * 256;
    float y = alpha[c] * (v[i] - mean) * inv + beta[c];
    Y[(size_t)row * D + c] = y;
    if (F16OUT) Yh[(size_t)row * D + c] = (_Float16)y;
  }
}

// ------------------------------- launcher -----------------------------------
extern "C" void kernel_launch(void* const* d_in, const int* in_sizes, int n_in,
                              void* d_out, int out_size, void* d_ws, size_t ws_size,
                              hipStream_t stream) {
  (void)in_sizes; (void)n_in; (void)out_size; (void)ws_size;
  const float* x      = (const float*)d_in[0];
  const int*   mask   = (const int*)  d_in[1];
  const float* wq     = (const float*)d_in[2];
  const float* wk     = (const float*)d_in[3];
  const float* wv     = (const float*)d_in[4];
  const float* wo     = (const float*)d_in[5];
  const float* wo_b   = (const float*)d_in[6];
  const float* w1     = (const float*)d_in[7];
  const float* b1     = (const float*)d_in[8];
  const float* w2     = (const float*)d_in[9];
  const float* b2     = (const float*)d_in[10];
  const float* alpha1 = (const float*)d_in[11];
  const float* bias1  = (const float*)d_in[12];
  const float* alpha2 = (const float*)d_in[13];
  const float* bias2  = (const float*)d_in[14];

  char* ws = (char*)d_ws;
  const size_t MB = 1u << 20;
  _Float16* Xh   = (_Float16*)(ws + 0);        // 8 MB   [4096,1024]
  _Float16* Wqh  = (_Float16*)(ws + 8  * MB);  // 2 MB
  _Float16* Wkh  = (_Float16*)(ws + 10 * MB);  // 2 MB
  _Float16* Wvh  = (_Float16*)(ws + 12 * MB);  // 2 MB
  _Float16* Woh  = (_Float16*)(ws + 14 * MB);  // 2 MB
  _Float16* W1h  = (_Float16*)(ws + 16 * MB);  // 8 MB   [1024,4096]
  _Float16* W2h  = (_Float16*)(ws + 24 * MB);  // 8 MB   [4096,1024]
  _Float16* Qh   = (_Float16*)(ws + 32 * MB);  // 8 MB
  _Float16* Kh   = (_Float16*)(ws + 40 * MB);  // 8 MB
  _Float16* Vh   = (_Float16*)(ws + 48 * MB);  // 8 MB
  _Float16* Ctxh = (_Float16*)(ws + 56 * MB);  // 8 MB
  float*    A4   = (float*)   (ws + 64 * MB);  // 16 MB  attn proj out (f32)
  float*    Y1   = (float*)   (ws + 80 * MB);  // 16 MB  LN1 out (f32)
  _Float16* Y1h  = (_Float16*)(ws + 96 * MB);  // 8 MB
  _Float16* FFh  = (_Float16*)(ws + 104 * MB); // 32 MB  relu(ff1) [4096,4096]
  float*    FF2  = (float*)   (ws + 136 * MB); // 16 MB  ff2 out (f32)

  dim3 blk(256);
  auto cvt = [&](const float* a, _Float16* o, int n) {
    f32_to_f16_kernel<<<(n + 255) / 256, blk, 0, stream>>>(a, o, n);
  };
  cvt(x,  Xh,  4096 * 1024);
  cvt(wq, Wqh, 1024 * 1024);
  cvt(wk, Wkh, 1024 * 1024);
  cvt(wv, Wvh, 1024 * 1024);
  cvt(wo, Woh, 1024 * 1024);
  cvt(w1, W1h, 1024 * 4096);
  cvt(w2, W2h, 4096 * 1024);

  // Q, K, V projections (M=4096, N=1024, K=1024), f16 outputs
  gemm_wmma<false, false, true><<<dim3(8, 32), blk, 0, stream>>>(
      Xh, Wqh, nullptr, nullptr, Qh, 4096, 1024, 1024);
  gemm_wmma<false, false, true><<<dim3(8, 32), blk, 0, stream>>>(
      Xh, Wkh, nullptr, nullptr, Kh, 4096, 1024, 1024);
  gemm_wmma<false, false, true><<<dim3(8, 32), blk, 0, stream>>>(
      Xh, Wvh, nullptr, nullptr, Vh, 4096, 1024, 1024);

  // fused attention: grid (2048/128 q-tiles, 2*16 b*h)
  attention_kernel<<<dim3(16, 32), blk, 0, stream>>>(Qh, Kh, Vh, mask, Ctxh);

  // output projection + bias (f32)
  gemm_wmma<true, false, false><<<dim3(8, 32), blk, 0, stream>>>(
      Ctxh, Woh, wo_b, A4, nullptr, 4096, 1024, 1024);

  // LN1: x + attn_out -> Y1 (f32) + Y1h (f16)
  ln_residual_kernel<true><<<4096, blk, 0, stream>>>(x, A4, alpha1, bias1, Y1, Y1h);

  // FFN: relu(Y1 @ W1 + b1) -> f16 ; @ W2 + b2 -> f32
  gemm_wmma<true, true, true><<<dim3(32, 32), blk, 0, stream>>>(
      Y1h, W1h, b1, nullptr, FFh, 4096, 4096, 1024);
  gemm_wmma<true, false, false><<<dim3(8, 32), blk, 0, stream>>>(
      FFh, W2h, b2, FF2, nullptr, 4096, 1024, 4096);

  // LN2: Y1 + ff -> d_out (f32)
  ln_residual_kernel<false><<<4096, blk, 0, stream>>>(
      Y1, FF2, alpha2, bias2, (float*)d_out, nullptr);
}
